// MonkeyJumpLinear_26113401159703
// MI455X (gfx1250) — matrix operand order repaired and checked by reading, not compiled, not run on hardware
//
#include <hip/hip_runtime.h>
#include <hip/hip_bf16.h>

typedef __bf16 bf16_t;
typedef __attribute__((ext_vector_type(4)))  __bf16 v4bf;
typedef __attribute__((ext_vector_type(8)))  __bf16 v8bf;
typedef __attribute__((ext_vector_type(16))) __bf16 v16bf;
typedef __attribute__((ext_vector_type(4)))  float  v4f;
typedef __attribute__((ext_vector_type(8)))  float  v8f;

#define M_TOK 8192   // B*T
#define N_H   2048
#define K_DIN 2048
#define RANK  8
#define TOPK  2
#define SCALING 2.0f

#define BM 128
#define BN 128
#define BK 32
#define LDK (BK + 8)          // bf16 elements per LDS row (padded) -> 80B row stride
#define NKT (K_DIN / BK)      // 64 k-tiles

// ---------------------------------------------------------------------------
// fp32 -> bf16 conversion (vectorized, one memory pass)
// ---------------------------------------------------------------------------
__global__ __launch_bounds__(256)
void cvt_f32_to_bf16(const float* __restrict__ src, bf16_t* __restrict__ dst, int n) {
  int i = (blockIdx.x * 256 + threadIdx.x) * 4;
  if (i >= n) return;
  v4f v = *(const v4f*)(src + i);
  v4bf o;
  o[0] = (__bf16)v[0]; o[1] = (__bf16)v[1]; o[2] = (__bf16)v[2]; o[3] = (__bf16)v[3];
  *(v4bf*)(dst + i) = o;
}

// ---------------------------------------------------------------------------
// per-token: mask = sum_k w*(id==expert);  xAs[m,r] = (x[m,:]·A[r,:]) * 2 * mask
// one wave32 per token
// ---------------------------------------------------------------------------
__global__ __launch_bounds__(256)
void lora_prep(const float* __restrict__ x, const float* __restrict__ A,
               const long long* __restrict__ ids, const float* __restrict__ w,
               const int* __restrict__ expertPtr, float* __restrict__ xAs) {
  int tok  = (blockIdx.x * 256 + threadIdx.x) >> 5;
  int lane = threadIdx.x & 31;
  if (tok >= M_TOK) return;

  const float* xr = x + (size_t)tok * K_DIN;
  float part[RANK];
#pragma unroll
  for (int r = 0; r < RANK; ++r) part[r] = 0.f;

  for (int d = lane; d < K_DIN; d += 32) {
    float xv = xr[d];
#pragma unroll
    for (int r = 0; r < RANK; ++r) part[r] += xv * A[r * K_DIN + d];
  }
#pragma unroll
  for (int r = 0; r < RANK; ++r) {
#pragma unroll
    for (int off = 16; off > 0; off >>= 1)
      part[r] += __shfl_xor(part[r], off, 32);
  }

  int expert = expertPtr[0];
  float mask = 0.f;
#pragma unroll
  for (int k = 0; k < TOPK; ++k)
    mask += w[tok * TOPK + k] * ((ids[tok * TOPK + k] == (long long)expert) ? 1.f : 0.f);

  if (lane < RANK) xAs[tok * RANK + lane] = part[lane] * (SCALING * mask);
}

// ---------------------------------------------------------------------------
// out = Xb(bf16) @ Wb(bf16)^T  (fp32 accum via v_wmma_f32_16x16x32_bf16)
//       + bias + xAs @ Blora^T   (fp32 epilogue)
// block tile 128x128, 8 waves (wave32), wave tile 64x32,
// double-buffered LDS filled with GLOBAL_LOAD_ASYNC_TO_LDS_B128 (ASYNCcnt)
// ---------------------------------------------------------------------------
__global__ __launch_bounds__(256)
void gemm_wmma(const bf16_t* __restrict__ Xb, const bf16_t* __restrict__ Wb,
               const float* __restrict__ bias, const float* __restrict__ xAs,
               const float* __restrict__ Blora, float* __restrict__ out) {
  __shared__ bf16_t As[2][BM * LDK];
  __shared__ bf16_t Bs[2][BN * LDK];
  __shared__ float  Xs[BM][RANK];

  const int tid    = threadIdx.x;
  const int lane   = tid & 31;
  const int waveId = tid >> 5;
  const int half   = (lane >> 4) & 1;   // which lane-half
  const int l16    = lane & 15;
  const int wm     = waveId >> 2;       // 0..1  (64-row slab)
  const int wn     = waveId & 3;        // 0..3  (32-col slab)
  const int mBlock = blockIdx.y * BM;
  const int nBlock = blockIdx.x * BN;

  // stage xAs rows of this M-block into LDS (128*8 floats, one v4f per thread)
  {
    const v4f* src = (const v4f*)(xAs + (size_t)mBlock * RANK);
    ((v4f*)&Xs[0][0])[tid] = src[tid];
  }

  // async global->LDS tile copy: 128 rows x 32 bf16 (64B/row) = 512 x 16B
  // chunks, 2 per thread, no register staging (ASYNCcnt-tracked)
  auto asyncTile = [&](const bf16_t* gbase, int kt, bf16_t* lbase) {
#pragma unroll
    for (int p = 0; p < 2; ++p) {
      int c   = tid + p * 256;
      int row = c >> 2;
      int col = (c & 3) * 8;
      const bf16_t* g = gbase + (size_t)row * K_DIN + kt * BK + col;
      unsigned lds    = (unsigned)(size_t)(lbase + row * LDK + col);
      asm volatile("global_load_async_to_lds_b128 %0, %1, off"
                   :: "v"(lds), "v"(g) : "memory");
    }
  };

  const bf16_t* Abase = Xb + (size_t)mBlock * K_DIN;
  const bf16_t* Bbase = Wb + (size_t)nBlock * K_DIN;

  v8f acc[4][2];
#pragma unroll
  for (int mt = 0; mt < 4; ++mt)
#pragma unroll
    for (int nt = 0; nt < 2; ++nt) acc[mt][nt] = (v8f)0.f;

  // prologue: fill buffer 0
  asyncTile(Abase, 0, &As[0][0]);
  asyncTile(Bbase, 0, &Bs[0][0]);

  for (int kt = 0; kt < NKT; ++kt) {
    const int buf = kt & 1;

    // my async writes into LDS[buf] have landed
    asm volatile("s_wait_asynccnt 0x0" ::: "memory");
    // everyone's writes landed; everyone done reading LDS[buf^1] (iter kt-1)
    __syncthreads();

    // kick off next tile into LDS[buf^1]; overlaps with WMMA below
    if (kt + 1 < NKT) {
      asyncTile(Abase, kt + 1, &As[buf ^ 1][0]);
      asyncTile(Bbase, kt + 1, &Bs[buf ^ 1][0]);
    }

    // fragment loads (per ISA VGPR layouts) + 8 WMMAs
    v16bf afrag[4], bfrag[2];
#pragma unroll
    for (int mt = 0; mt < 4; ++mt) {
      // A 16x32 bf16: lane-half 0 -> K 0..7 & 16..23 ; half 1 -> K 8..15 & 24..31
      const bf16_t* p = &As[buf][(wm * 64 + mt * 16 + l16) * LDK + half * 8];
      v8bf c0 = *(const v8bf*)p;
      v8bf c1 = *(const v8bf*)(p + 16);
      afrag[mt] = __builtin_shufflevector(c0, c1, 0,1,2,3,4,5,6,7,8,9,10,11,12,13,14,15);
    }
#pragma unroll
    for (int nt = 0; nt < 2; ++nt) {
      // B 32x16 bf16: lane-half 0 -> K 0..7 & 16..23 ; half 1 -> K 8..15 & 24..31
      const bf16_t* p = &Bs[buf][(wn * 32 + nt * 16 + l16) * LDK + half * 8];
      v8bf c0 = *(const v8bf*)p;
      v8bf c1 = *(const v8bf*)(p + 16);
      bfrag[nt] = __builtin_shufflevector(c0, c1, 0,1,2,3,4,5,6,7,8,9,10,11,12,13,14,15);
    }
#pragma unroll
    for (int mt = 0; mt < 4; ++mt)
#pragma unroll
      for (int nt = 0; nt < 2; ++nt)
        acc[mt][nt] = __builtin_amdgcn_wmma_f32_16x16x32_bf16(
            false, afrag[mt], false, bfrag[nt], (short)0, acc[mt][nt], false, false);
  }

  // epilogue: + bias + rank-8 LoRA correction (mask/scaling pre-folded into xAs)
#pragma unroll
  for (int nt = 0; nt < 2; ++nt) {
    int n = nBlock + wn * 32 + nt * 16 + l16;
    float bn = bias[n];
    v4f bl0 = *(const v4f*)(Blora + (size_t)n * RANK);
    v4f bl1 = *(const v4f*)(Blora + (size_t)n * RANK + 4);
#pragma unroll
    for (int mt = 0; mt < 4; ++mt) {
      int mrow = wm * 64 + mt * 16 + half * 8;    // C layout: VGPR v -> M = v + 8*half
#pragma unroll
      for (int v = 0; v < 8; ++v) {
        int ml = mrow + v;
        v4f xa0 = *(const v4f*)&Xs[ml][0];
        v4f xa1 = *(const v4f*)&Xs[ml][4];
        float d = bn
                + xa0[0]*bl0[0] + xa0[1]*bl0[1] + xa0[2]*bl0[2] + xa0[3]*bl0[3]
                + xa1[0]*bl1[0] + xa1[1]*bl1[1] + xa1[2]*bl1[2] + xa1[3]*bl1[3];
        out[(size_t)(mBlock + ml) * N_H + n] = acc[mt][nt][v] + d;
      }
    }
  }
}

// ---------------------------------------------------------------------------
extern "C" void kernel_launch(void* const* d_in, const int* in_sizes, int n_in,
                              void* d_out, int out_size, void* d_ws, size_t ws_size,
                              hipStream_t stream) {
  const float*      x   = (const float*)d_in[0];
  const float*      W   = (const float*)d_in[1];
  const float*      b   = (const float*)d_in[2];
  const float*      A   = (const float*)d_in[3];
  const float*      Bl  = (const float*)d_in[4];
  const long long*  ids = (const long long*)d_in[5];
  const float*      tkw = (const float*)d_in[6];
  const int*        eid = (const int*)d_in[7];
  float*            out = (float*)d_out;

  // workspace layout
  char* ws = (char*)d_ws;
  bf16_t* Xb  = (bf16_t*)ws;                                   // 32 MB
  bf16_t* Wb  = (bf16_t*)(ws + (size_t)M_TOK * K_DIN * 2);     //  8 MB
  float*  xAs = (float*)(ws + (size_t)M_TOK * K_DIN * 2
                            + (size_t)N_H * K_DIN * 2);        // 256 KB

  // 1) downconvert x and W to bf16
  {
    int n = M_TOK * K_DIN;
    cvt_f32_to_bf16<<<n / (256 * 4), 256, 0, stream>>>(x, Xb, n);
  }
  {
    int n = N_H * K_DIN;
    cvt_f32_to_bf16<<<n / (256 * 4), 256, 0, stream>>>(W, Wb, n);
  }

  // 2) routing mask + rank-8 projection (scaling*mask folded in)
  lora_prep<<<M_TOK / 8, 256, 0, stream>>>(x, A, ids, tkw, eid, xAs);

  // 3) WMMA GEMM with fused bias + LoRA epilogue
  dim3 grid(N_H / BN, M_TOK / BM);
  gemm_wmma<<<grid, 256, 0, stream>>>(Xb, Wb, b, xAs, Bl, out);
}
